// FRIEND_67680094650645
// MI455X (gfx1250) — matrix operand order repaired and checked by reading, not compiled, not run on hardware
//
#include <hip/hip_runtime.h>
#include <hip/hip_bf16.h>

typedef __bf16 bf16;
typedef __attribute__((ext_vector_type(16))) __bf16 v16bf;
typedef __attribute__((ext_vector_type(8)))  __bf16 v8bf;
typedef __attribute__((ext_vector_type(4)))  __bf16 v4bf;
typedef __attribute__((ext_vector_type(8)))  float  v8f;

#define HD    128
#define H3    384
#define LFQ   64
#define LCQ   256
#define TILE  16
#define FMAXQ 16
#define CH    32        // rows of common_x staged per async chunk

__device__ __forceinline__ float sigmoidf_(float x) { return 1.0f / (1.0f + expf(-x)); }

// A fragment (16x32 bf16): one contiguous 16-element run per lane (k = 16*half + s).
__device__ __forceinline__ v16bf load_a_frag(const bf16* base) {
    v8bf lo = *(const v8bf*)(base);
    v8bf hi = *(const v8bf*)(base + 8);
    return __builtin_shufflevector(lo, hi, 0,1,2,3,4,5,6,7,8,9,10,11,12,13,14,15);
}

__device__ __forceinline__ v4bf cvt4(float4 v) {
    v4bf p = { (bf16)v.x, (bf16)v.y, (bf16)v.z, (bf16)v.w };
    return p;
}

// ---------------------------------------------------------------------------
// Kernel 1: GRU over friend_x, 16 sequences per workgroup, 64 sequential steps.
// Weights LDS-resident in bf16 (2 x 96KB; CDNA5 WGP has 320KB LDS).
// Per step: gi = x_t @ W_ih^T and gh = h @ W_hh^T via v_wmma_f32_16x16x32_bf16
// with f32 accumulate; gates + hidden state in f32; snapshot at last valid t.
// ---------------------------------------------------------------------------
extern "C" __global__ __launch_bounds__(256)
void gru_friend_kernel(const float* __restrict__ fx,      // [N][LF][HD]
                       const float* __restrict__ W_ih,    // [H3][HD]
                       const float* __restrict__ W_hh,    // [H3][HD]
                       const float* __restrict__ b_ih,    // [H3]
                       const float* __restrict__ b_hh,    // [H3]
                       const int*   __restrict__ fmask,   // [N][LF]
                       float*       __restrict__ fout,    // [N][HD]
                       int N)
{
    extern __shared__ char smem[];
    bf16*  w_ih_l = (bf16*)smem;                       // [H3][HD] bf16  (96 KB)
    bf16*  w_hh_l = w_ih_l + H3 * HD;                  // [H3][HD] bf16  (96 KB)
    float* gi     = (float*)(w_hh_l + H3 * HD);        // [TILE][H3]     (24 KB)
    float* gh     = gi + TILE * H3;                    // [TILE][H3]     (24 KB)
    bf16*  x_bf   = (bf16*)(gh + TILE * H3);           // [TILE][HD]     (4 KB)
    bf16*  h_bf   = x_bf + TILE * HD;                  // [TILE][HD]     (4 KB)
    float* h_f    = (float*)(h_bf + TILE * HD);        // [TILE][HD]     (8 KB)
    float* b_r    = h_f + TILE * HD;                   // [HD] bias tables (2 KB)
    float* b_z    = b_r + HD;
    float* b_in   = b_z + HD;
    float* b_hn   = b_in + HD;
    int*   ilast  = (int*)(b_hn + HD);                 // [TILE]

    const int tid  = threadIdx.x;
    const int lane = tid & 31;
    const int wave = tid >> 5;
    const int m16  = lane & 15;
    const int half = lane >> 4;
    const int row0 = blockIdx.x * TILE;

    // Stage weights (one-time): f32 global -> bf16 LDS, [n_out][k] (B-friendly).
    for (int i4 = tid; i4 < (H3 * HD) / 4; i4 += 256) {
        float4 a = *(const float4*)&W_ih[i4 * 4];
        float4 b = *(const float4*)&W_hh[i4 * 4];
        *(v4bf*)&w_ih_l[i4 * 4] = cvt4(a);
        *(v4bf*)&w_hh_l[i4 * 4] = cvt4(b);
    }
    for (int i = tid; i < TILE * HD; i += 256) {
        h_f[i]  = 0.0f;
        h_bf[i] = (bf16)0.0f;
    }
    if (tid < HD) {
        b_r[tid]  = b_ih[tid] + b_hh[tid];
        b_z[tid]  = b_ih[HD + tid] + b_hh[HD + tid];
        b_in[tid] = b_ih[2 * HD + tid];
        b_hn[tid] = b_hh[2 * HD + tid];
    }
    if (tid < TILE) {
        int r = min(row0 + tid, N - 1);
        const int* mrow = fmask + (size_t)r * LFQ;
        int cnt = 0;
        for (int t2 = 0; t2 < LFQ; ++t2) cnt += mrow[t2];
        ilast[tid] = cnt - 1;
    }
    __syncthreads();

    for (int t = 0; t < LFQ; ++t) {
        // stage x_t (16 rows x 128) -> bf16 LDS, vectorized (2 b128 loads/thread)
        for (int i4 = tid; i4 < (TILE * HD) / 4; i4 += 256) {
            int s = i4 >> 5, c4 = (i4 & 31) * 4;
            int r = min(row0 + s, N - 1);
            float4 v = *(const float4*)&fx[((size_t)r * LFQ + t) * HD + c4];
            *(v4bf*)&x_bf[s * HD + c4] = cvt4(v);
        }
        // prefetch next step's slab, one thread per 128B cacheline, near scope
        if (tid < 64 && (t + 1) < LFQ) {
            int s = tid >> 2, ln = tid & 3;
            int r = min(row0 + s, N - 1);
            __builtin_prefetch(&fx[((size_t)r * LFQ + (t + 1)) * HD + ln * 32], 0, 3);
        }
        __syncthreads();

        // A fragments (16x32 bf16) for x_t and h over the 4 K-chunks.
        v16bf ax[4], ah[4];
        #pragma unroll
        for (int kc = 0; kc < 4; ++kc) {
            const int off = m16 * HD + kc * 32 + 16 * half;
            ax[kc] = load_a_frag(x_bf + off);
            ah[kc] = load_a_frag(h_bf + off);
        }

        // Each wave owns 3 of the 24 output n-tiles (384/16) for BOTH gemms.
        #pragma unroll
        for (int j = 0; j < 3; ++j) {
            const int nt = wave * 3 + j;
            const int ng = nt * 16 + m16;      // output column for this lane
            v8f accI = {}, accH = {};
            #pragma unroll
            for (int kc = 0; kc < 4; ++kc) {
                // B fragment = one contiguous 32B run in LDS.
                v16bf bi = *(const v16bf*)(w_ih_l + ng * HD + kc * 32 + 16 * half);
                v16bf bh = *(const v16bf*)(w_hh_l + ng * HD + kc * 32 + 16 * half);
                accI = __builtin_amdgcn_wmma_f32_16x16x32_bf16(
                           false, ax[kc], false, bi, (short)0, accI, false, false);
                accH = __builtin_amdgcn_wmma_f32_16x16x32_bf16(
                           false, ah[kc], false, bh, (short)0, accH, false, false);
            }
            #pragma unroll
            for (int v = 0; v < 8; ++v) {
                int m = v + 8 * half;
                gi[m * H3 + ng] = accI[v];
                gh[m * H3 + ng] = accH[v];
            }
        }
        __syncthreads();

        // Gate math in f32: each thread owns 8 CONSECUTIVE columns of one row,
        // so all gi/gh/h LDS traffic and the fout store merge into b128 ops.
        {
            const int s  = tid >> 4;
            const int c0 = (tid & 15) << 3;
            const bool snap = (t == ilast[s]) && (row0 + s) < N;
            #pragma unroll
            for (int q = 0; q < 8; ++q) {
                const int c = c0 + q;
                float rg = sigmoidf_(gi[s*H3 + c]      + gh[s*H3 + c]      + b_r[c]);
                float zg = sigmoidf_(gi[s*H3 + HD + c] + gh[s*H3 + HD + c] + b_z[c]);
                float nn = tanhf(gi[s*H3 + 2*HD + c] + b_in[c]
                                 + rg * (gh[s*H3 + 2*HD + c] + b_hn[c]));
                float hn = (1.0f - zg) * nn + zg * h_f[s * HD + c];
                h_f[s * HD + c]  = hn;
                h_bf[s * HD + c] = (bf16)hn;
                if (snap) fout[(size_t)(row0 + s) * HD + c] = hn;
            }
        }
        __syncthreads();
    }
}

// ---------------------------------------------------------------------------
// Kernel 2: sf = [self_rep ; friend_out] @ W_friend^T   ([16,256]x[256->128])
//           u  = sf @ W_beta                            ([16,128]x[128->128])
// bf16 WMMA, f32 accumulate. (W_beta folded: scores[n,l] = common_x[n,l,:].u[n])
// ---------------------------------------------------------------------------
extern "C" __global__ __launch_bounds__(256)
void sf_u_kernel(const float* __restrict__ self_x,   // [B][HD]
                 const float* __restrict__ fout,     // [N][HD]
                 const float* __restrict__ W_friend, // [HD][2*HD]
                 const float* __restrict__ W_beta,   // [HD][HD]
                 const int*   __restrict__ user_idx, // [N]
                 float*       __restrict__ u,        // [N][HD]
                 int N)
{
    __shared__ bf16  cat_bf[TILE * 2 * HD];  // 8 KB
    __shared__ float sf_f[TILE * HD];        // 8 KB
    __shared__ bf16  sf_bf[TILE * HD];       // 4 KB

    const int tid  = threadIdx.x;
    const int lane = tid & 31;
    const int wave = tid >> 5;
    const int m16  = lane & 15;
    const int half = lane >> 4;
    const int row0 = blockIdx.x * TILE;

    for (int i4 = tid; i4 < (TILE * 2 * HD) / 4; i4 += 256) {
        int s = i4 >> 6, c4 = (i4 & 63) * 4;
        int r = min(row0 + s, N - 1);
        float4 v = (c4 < HD)
            ? *(const float4*)&self_x[(size_t)user_idx[r] * HD + c4]
            : *(const float4*)&fout[(size_t)r * HD + (c4 - HD)];
        *(v4bf*)&cat_bf[s * (2 * HD) + c4] = cvt4(v);
    }
    __syncthreads();

    // sf: each wave owns one of 8 n-tiles; 8 K-chunks of 32 over 256.
    {
        const int ng = wave * 16 + m16;
        v8f acc = {};
        #pragma unroll
        for (int kc = 0; kc < 8; ++kc) {
            v16bf a = load_a_frag(cat_bf + m16 * (2 * HD) + kc * 32 + 16 * half);
            v16bf b;
            const float* wp = W_friend + (size_t)ng * (2 * HD) + kc * 32 + 16 * half;
            #pragma unroll
            for (int e = 0; e < 16; ++e) b[e] = (bf16)wp[e];
            acc = __builtin_amdgcn_wmma_f32_16x16x32_bf16(
                      false, a, false, b, (short)0, acc, false, false);
        }
        #pragma unroll
        for (int v = 0; v < 8; ++v) {
            int m = v + 8 * half;
            sf_f[m * HD + ng] = acc[v];
        }
    }
    __syncthreads();
    for (int i = tid; i < TILE * HD; i += 256) sf_bf[i] = (bf16)sf_f[i];
    __syncthreads();

    // u = sf @ W_beta :  B[k][n] = W_beta[k*HD + n] (strided gather + cvt).
    {
        const int ng = wave * 16 + m16;
        v8f acc = {};
        #pragma unroll
        for (int kc = 0; kc < 4; ++kc) {
            v16bf a = load_a_frag(sf_bf + m16 * HD + kc * 32 + 16 * half);
            v16bf b;
            #pragma unroll
            for (int e = 0; e < 16; ++e)
                b[e] = (bf16)W_beta[(size_t)(kc * 32 + 16 * half + e) * HD + ng];
            acc = __builtin_amdgcn_wmma_f32_16x16x32_bf16(
                      false, a, false, b, (short)0, acc, false, false);
        }
        #pragma unroll
        for (int v = 0; v < 8; ++v) {
            int m = v + 8 * half;
            if ((row0 + m) < N)
                u[(size_t)(row0 + m) * HD + ng] = acc[v];
        }
    }
}

// ---------------------------------------------------------------------------
// Kernel 3: tf[n] = sum_l softplus(common_x[n,l,:].u[n]) * exp(-t/TAU+1) * mask
// The 570 MB streaming pass (the roofline floor). common_x is staged to LDS
// with CDNA5 async copies (global_load_async_to_lds_b128, ASYNCcnt) in
// double-buffered 16 KB chunks; dots run out of LDS, 8 lanes per row so every
// global b128 issue is fully cacheline-coalesced.
// ---------------------------------------------------------------------------
extern "C" __global__ __launch_bounds__(256)
void tf_kernel(const float* __restrict__ common_x,    // [N][LC][HD]
               const float* __restrict__ common_time, // [N][LC]
               const int*   __restrict__ cmask,       // [N][LC]
               const float* __restrict__ u,           // [N][HD]
               float*       __restrict__ tf,          // [N]
               int N)
{
    __shared__ __align__(16) float stage[2][CH * HD]; // 2 x 16 KB
    __shared__ float u_l[HD];
    __shared__ float partial[8];

    const int n   = blockIdx.x;
    const int tid = threadIdx.x;
    if (tid < HD) u_l[tid] = u[(size_t)n * HD + tid];

    const uint64_t gbase = (uint64_t)(uintptr_t)(common_x + (size_t)n * LCQ * HD);
    const uint32_t lds0  = (uint32_t)(uintptr_t)&stage[0][0];
    const uint32_t lds1  = (uint32_t)(uintptr_t)&stage[1][0];

    // kick off chunk 0 (each wave issues 4 async b128 ops -> ASYNCcnt=4)
    #pragma unroll
    for (int q = 0; q < 4; ++q) {
        uint32_t off = (uint32_t)(q * 256 + tid) * 16u;
        uint32_t la  = lds0 + off;
        asm volatile("global_load_async_to_lds_b128 %0, %1, %2"
                     :: "v"(la), "v"(off), "s"(gbase) : "memory");
    }

    const int r   = tid >> 3;   // row within chunk (0..31)
    const int seg = tid & 7;    // 8 lanes cooperate on one row
    float acc = 0.0f;

    for (int c = 0; c < LCQ / CH; ++c) {
        const int buf = c & 1;
        if (c + 1 < LCQ / CH) {
            const uint32_t ldsn  = buf ? lds0 : lds1;
            const uint32_t goff0 = (uint32_t)((c + 1) * CH * HD) * 4u;
            #pragma unroll
            for (int q = 0; q < 4; ++q) {
                uint32_t eo  = (uint32_t)(q * 256 + tid) * 16u;
                uint32_t off = goff0 + eo;
                uint32_t la  = ldsn + eo;
                asm volatile("global_load_async_to_lds_b128 %0, %1, %2"
                             :: "v"(la), "v"(off), "s"(gbase) : "memory");
            }
            asm volatile("s_wait_asynccnt 0x4" ::: "memory"); // chunk c landed
        } else {
            asm volatile("s_wait_asynccnt 0x0" ::: "memory");
        }
        __syncthreads();

        const float* rowp = &stage[buf][r * HD];
        float dot = 0.0f;
        #pragma unroll
        for (int q2 = 0; q2 < 4; ++q2) {
            const float4 v = ((const float4*)rowp)[seg + 8 * q2];
            const int k = 4 * (seg + 8 * q2);
            dot += v.x * u_l[k] + v.y * u_l[k+1] + v.z * u_l[k+2] + v.w * u_l[k+3];
        }
        dot += __shfl_down(dot, 4, 8);
        dot += __shfl_down(dot, 2, 8);
        dot += __shfl_down(dot, 1, 8);
        if (seg == 0) {
            const int l = c * CH + r;
            float sp = (dot > 0.0f) ? (dot + log1pf(expf(-dot))) : log1pf(expf(dot));
            float tw = expf(-common_time[(size_t)n * LCQ + l] * 1e-6f + 1.0f);
            acc += sp * tw * (float)cmask[(size_t)n * LCQ + l];
        }
        __syncthreads();   // all lanes done with buf before it is refilled
    }

    #pragma unroll
    for (int off = 16; off > 0; off >>= 1) acc += __shfl_down(acc, off, 32);
    if ((tid & 31) == 0) partial[tid >> 5] = acc;
    __syncthreads();
    if (tid == 0) {
        float s = 0.0f;
        #pragma unroll
        for (int w = 0; w < 8; ++w) s += partial[w];
        tf[n] = s;
    }
}

// ---------------------------------------------------------------------------
// Kernel 4: pad to groups, softmax over FMAX (incl. padded zeros, faithful),
// weighted aggregation -> out [B][HD]. One block per group, thread per h.
// ---------------------------------------------------------------------------
extern "C" __global__ __launch_bounds__(128)
void agg_kernel(const float* __restrict__ fout,       // [N][HD]
                const float* __restrict__ tf,         // [N]
                const int*   __restrict__ gather_idx, // [B][FMAX]
                const int*   __restrict__ pad_mask,   // [B][FMAX]
                float*       __restrict__ out)        // [B][HD]
{
    __shared__ float tfp[FMAXQ];
    __shared__ int   gidx[FMAXQ];
    __shared__ float pm_s[FMAXQ];
    const int b   = blockIdx.x;
    const int tid = threadIdx.x;
    if (tid < FMAXQ) {
        int g    = gather_idx[b * FMAXQ + tid];
        float pm = (float)pad_mask[b * FMAXQ + tid];
        gidx[tid] = g;
        pm_s[tid] = pm;
        tfp[tid]  = tf[g] * pm;
    }
    __syncthreads();

    float mx = -3.402823e38f;
    #pragma unroll
    for (int f = 0; f < FMAXQ; ++f) mx = fmaxf(mx, tfp[f]);
    float ex[FMAXQ];
    float den = 0.0f;
    #pragma unroll
    for (int f = 0; f < FMAXQ; ++f) { ex[f] = expf(tfp[f] - mx); den += ex[f]; }
    float inv = 1.0f / den;

    float acc = 0.0f;
    #pragma unroll
    for (int f = 0; f < FMAXQ; ++f)
        acc += (ex[f] * inv) * pm_s[f] * fout[(size_t)gidx[f] * HD + tid];
    out[(size_t)b * HD + tid] = acc;
}

// ---------------------------------------------------------------------------
extern "C" void kernel_launch(void* const* d_in, const int* in_sizes, int n_in,
                              void* d_out, int out_size, void* d_ws, size_t ws_size,
                              hipStream_t stream)
{
    const float* self_x      = (const float*)d_in[0];
    const float* common_x    = (const float*)d_in[1];
    const float* common_time = (const float*)d_in[2];
    const float* friend_x    = (const float*)d_in[3];
    const float* W_ih        = (const float*)d_in[4];
    const float* W_hh        = (const float*)d_in[5];
    const float* b_ih        = (const float*)d_in[6];
    const float* b_hh        = (const float*)d_in[7];
    const float* W_friend    = (const float*)d_in[8];
    const float* W_beta      = (const float*)d_in[9];
    const int*   fmask       = (const int*)d_in[10];
    const int*   cmask       = (const int*)d_in[11];
    const int*   user_idx    = (const int*)d_in[12];
    const int*   gather_idx  = (const int*)d_in[13];
    const int*   pad_mask    = (const int*)d_in[14];
    float* out = (float*)d_out;

    const int N  = in_sizes[12];          // 4352
    const int Bq = in_sizes[13] / FMAXQ;  // 512

    float* fout = (float*)d_ws;                        // [N][HD]
    float* u    = fout + (size_t)N * HD;               // [N][HD]
    float* tf   = u    + (size_t)N * HD;               // [N]

    const int nTiles = (N + TILE - 1) / TILE;          // 272

    const size_t gru_lds =
        (size_t)2 * H3 * HD * sizeof(bf16)    // weights bf16
      + (size_t)2 * TILE * H3 * sizeof(float) // gi, gh
      + (size_t)2 * TILE * HD * sizeof(bf16)  // x_bf, h_bf
      + (size_t)TILE * HD * sizeof(float)     // h_f
      + (size_t)4 * HD * sizeof(float)        // bias tables
      + (size_t)TILE * sizeof(int);           // ilast (~258 KB of the 320 KB WGP LDS)
    hipFuncSetAttribute((const void*)gru_friend_kernel,
                        hipFuncAttributeMaxDynamicSharedMemorySize, (int)gru_lds);

    gru_friend_kernel<<<nTiles, 256, gru_lds, stream>>>(
        friend_x, W_ih, W_hh, b_ih, b_hh, fmask, fout, N);
    sf_u_kernel<<<nTiles, 256, 0, stream>>>(
        self_x, fout, W_friend, W_beta, user_idx, u, N);
    tf_kernel<<<N, 256, 0, stream>>>(
        common_x, common_time, cmask, u, tf, N);
    agg_kernel<<<Bq, 128, 0, stream>>>(
        fout, tf, gather_idx, pad_mask, out);
}
